// Attention_18081812316820
// MI455X (gfx1250) — matrix operand order repaired and checked by reading, not compiled, not run on hardware
//
#include <hip/hip_runtime.h>

// ---------------------------------------------------------------------------
// Problem constants (match the JAX reference)
// ---------------------------------------------------------------------------
#define BATCH   2
#define SEQ     2048
#define DIMC    1024
#define NHEADS  16
#define HDIM    64
#define ROWS    (BATCH * SEQ)          // 4096
#define QKV_O   (3 * DIMC)             // 3072
#define LN2_10K 0.28782313662425572f   // ln(10000)/32

typedef __attribute__((ext_vector_type(16))) __bf16 v16bf;
typedef __attribute__((ext_vector_type(8)))  __bf16 v8bf;
typedef __attribute__((ext_vector_type(8)))  float  v8f;
typedef __attribute__((ext_vector_type(4)))  int    v4i;

typedef __attribute__((address_space(1))) v4i* gv4i_p;   // global 16B vector ptr
typedef __attribute__((address_space(3))) v4i* lv4i_p;   // LDS    16B vector ptr

#define WMMA_BF16(a, b, c) \
    __builtin_amdgcn_wmma_f32_16x16x32_bf16(false, (a), false, (b), (short)0, (c), false, false)

// ---------------------------------------------------------------------------
// Async global -> LDS copy (16 bytes / lane), ASYNCcnt-tracked.
// Builtin signature (probed from this toolchain's diagnostic):
//   void(v4i addrspace(1)* gsrc, v4i addrspace(3)* ldst, imm i32 off, imm i32 cpol)
// ---------------------------------------------------------------------------
#if __has_builtin(__builtin_amdgcn_global_load_async_to_lds_b128)
#define HAVE_ASYNC_LDS 1
#else
#define HAVE_ASYNC_LDS 0
#endif

__device__ __forceinline__ void cp16_async(__bf16* lds, const __bf16* g) {
#if HAVE_ASYNC_LDS
    __builtin_amdgcn_global_load_async_to_lds_b128((gv4i_p)g, (lv4i_p)lds, 0, 0);
#else
    *(v8bf*)lds = *(const v8bf*)g;
#endif
}

#if HAVE_ASYNC_LDS
#define WAIT_ASYNC(n) asm volatile("s_wait_asynccnt %0" ::"i"(n) : "memory")
#else
#define WAIT_ASYNC(n)
#endif

// Load a 16x32 bf16 A-fragment: lane holds row (lane&15),
// elements 0..15 = K (half*16 .. half*16+15), contiguous.
// `p` must already point at row_base + half*16.
__device__ __forceinline__ v16bf load_a16x32(const __bf16* p) {
    v8bf c0 = *(const v8bf*)(p);
    v8bf c1 = *(const v8bf*)(p + 8);
    v16bf a;
#pragma unroll
    for (int i = 0; i < 8; ++i) { a[i] = c0[i]; a[i + 8] = c1[i]; }
    return a;
}

// ---------------------------------------------------------------------------
// Kernel 0: f32 -> bf16 conversion
// ---------------------------------------------------------------------------
__global__ void cvt_f32_to_bf16(const float* __restrict__ in, __bf16* __restrict__ out, int n) {
    int i = blockIdx.x * blockDim.x + threadIdx.x;
    if (i < n) out[i] = (__bf16)in[i];
}

// ---------------------------------------------------------------------------
// Kernel 1/4: WMMA bf16 GEMM  C[M,N] = A[M,K] * Bw[N,K]^T (+ bias)
// Block (8 waves) computes a 64x128 tile; per-32-K slabs of A (64x32) and
// B (128x32) are double-buffered in LDS, filled with async global->LDS
// copies and consumed as WMMA fragments via ds loads.
// ---------------------------------------------------------------------------
__global__ __launch_bounds__(256) void gemm_bf16_wmma(
    const __bf16* __restrict__ A, const __bf16* __restrict__ Bw,
    float* __restrict__ C, const float* __restrict__ bias,
    int M, int N, int K)
{
    __shared__ __attribute__((aligned(32))) __bf16 As[2][64][32];    // 2 x 4 KB
    __shared__ __attribute__((aligned(32))) __bf16 Bs[2][128][32];   // 2 x 8 KB

    const int tid   = threadIdx.x;
    const int lane  = tid & 31;
    const int half  = lane >> 4;
    const int lr    = lane & 15;
    const int wave  = tid >> 5;
    const int waveM = wave >> 1;   // 0..3  -> 16-row slice
    const int waveN = wave & 1;    // 0..1  -> 64-col slice

    const int tilesN = N >> 7;
    const int m0 = (blockIdx.x / tilesN) * 64;
    const int n0 = (blockIdx.x % tilesN) * 128;

    // per-thread stage-copy assignment (3 x 16B per thread per 32-K slab)
    const int ar = tid >> 2, ac = (tid & 3) * 8;           // A: 64 rows x 4 chunks
    const int br0 = tid >> 2, bc0 = (tid & 3) * 8;         // B rows 0..63
    const int br1 = (tid + 256) >> 2, bc1 = (tid & 3) * 8; // B rows 64..127

    auto issue_stage = [&](int k0, int buf) {
        cp16_async(&As[buf][ar][ac],   A  + (size_t)(m0 + ar)  * K + k0 + ac);
        cp16_async(&Bs[buf][br0][bc0], Bw + (size_t)(n0 + br0) * K + k0 + bc0);
        cp16_async(&Bs[buf][br1][bc1], Bw + (size_t)(n0 + br1) * K + k0 + bc1);
    };

    v8f acc[4] = {};
    const int stages = K >> 5;
    issue_stage(0, 0);

    for (int it = 0; it < stages; ++it) {
        const int buf = it & 1;
        if (it + 1 < stages) {
            issue_stage((it + 1) << 5, buf ^ 1);
            WAIT_ASYNC(3);        // own stage-`it` copies done; next stage in flight
        } else {
            WAIT_ASYNC(0);
        }
        __syncthreads();          // all waves' copies for this slab published

        const v16bf a = load_a16x32(&As[buf][waveM * 16 + lr][half * 16]);
#pragma unroll
        for (int c = 0; c < 4; ++c) {
            const v16bf b = *(const v16bf*)&Bs[buf][waveN * 64 + c * 16 + lr][half * 16];
            acc[c] = WMMA_BF16(a, b, acc[c]);
        }
        __syncthreads();          // done reading this buffer before it is refilled
    }

    const int mw = m0 + waveM * 16;
    const int nw = n0 + waveN * 64;
#pragma unroll
    for (int j = 0; j < 8; ++j) {
        const int row = mw + j + half * 8;   // C layout: VGPR j -> M = j (+8 upper half)
#pragma unroll
        for (int c = 0; c < 4; ++c) {
            const int col = nw + c * 16 + lr;
            float v = acc[c][j];
            if (bias) v += bias[col];
            C[(size_t)row * N + col] = v;
        }
    }
}

// ---------------------------------------------------------------------------
// Kernel 2: per-head LayerNorm + RoPE on q,k; repack q,k -> [B,H,S,D] bf16,
// v -> transposed [B,H,D,S] bf16 (so P*V WMMA B-fragments read contiguous keys).
// One wave per (b,s,h); each lane owns d = lane and d = lane+32.
// ---------------------------------------------------------------------------
__global__ __launch_bounds__(256) void lnrope_pack(
    const float* __restrict__ qkv,
    const float* __restrict__ qg, const float* __restrict__ qbta,
    const float* __restrict__ kg, const float* __restrict__ kbta,
    __bf16* __restrict__ qo, __bf16* __restrict__ ko, __bf16* __restrict__ vt)
{
    const int wave = threadIdx.x >> 5;
    const int lane = threadIdx.x & 31;
    const int row  = blockIdx.x * 8 + wave;        // 0 .. B*S*H-1
    const int h = row & (NHEADS - 1);
    const int s = (row >> 4) & (SEQ - 1);
    const int b = row >> 15;

    const size_t base = ((size_t)(b * SEQ + s)) * QKV_O + h * HDIM;
    const int d0 = lane, d1 = lane + 32;

    const float invf = __expf(-(float)lane * LN2_10K);
    const float ang  = (float)s * invf;
    const float cs = __cosf(ang), sn = __sinf(ang);

    const size_t oq = (((size_t)(b * NHEADS + h)) * SEQ + s) * HDIM;

    // ---- q: LN + RoPE + scale
    {
        float x0 = qkv[base + d0], x1 = qkv[base + d1];
        float sum = x0 + x1;
#pragma unroll
        for (int m = 16; m >= 1; m >>= 1) sum += __shfl_xor(sum, m, 32);
        const float mu = sum * (1.0f / 64.0f);
        float e0 = x0 - mu, e1 = x1 - mu;
        float vs = e0 * e0 + e1 * e1;
#pragma unroll
        for (int m = 16; m >= 1; m >>= 1) vs += __shfl_xor(vs, m, 32);
        const float rstd = rsqrtf(vs * (1.0f / 64.0f) + 1e-6f);
        const float y0 = e0 * rstd * qg[d0] + qbta[d0];
        const float y1 = e1 * rstd * qg[d1] + qbta[d1];
        qo[oq + d0] = (__bf16)((y0 * cs - y1 * sn) * 0.125f);  // * D^-0.5
        qo[oq + d1] = (__bf16)((y1 * cs + y0 * sn) * 0.125f);
    }
    // ---- k: LN + RoPE
    {
        float x0 = qkv[base + DIMC + d0], x1 = qkv[base + DIMC + d1];
        float sum = x0 + x1;
#pragma unroll
        for (int m = 16; m >= 1; m >>= 1) sum += __shfl_xor(sum, m, 32);
        const float mu = sum * (1.0f / 64.0f);
        float e0 = x0 - mu, e1 = x1 - mu;
        float vs = e0 * e0 + e1 * e1;
#pragma unroll
        for (int m = 16; m >= 1; m >>= 1) vs += __shfl_xor(vs, m, 32);
        const float rstd = rsqrtf(vs * (1.0f / 64.0f) + 1e-6f);
        const float y0 = e0 * rstd * kg[d0] + kbta[d0];
        const float y1 = e1 * rstd * kg[d1] + kbta[d1];
        ko[oq + d0] = (__bf16)(y0 * cs - y1 * sn);
        ko[oq + d1] = (__bf16)(y1 * cs + y0 * sn);
    }
    // ---- v: transpose to [B,H,D,S]
    {
        const size_t ob = ((size_t)(b * NHEADS + h)) * HDIM;
        vt[(ob + d0) * SEQ + s] = (__bf16)qkv[base + 2 * DIMC + d0];
        vt[(ob + d1) * SEQ + s] = (__bf16)qkv[base + 2 * DIMC + d1];
    }
}

// ---------------------------------------------------------------------------
// Kernel 3: flash attention. Block = 4 waves sharing one (b,h); each wave
// owns 16 queries. K (32x64) and V (64x32) tiles are double-buffered in LDS
// via async copies and shared by all 4 waves. 8 WMMAs per 32-key tile.
// P moves C-layout -> A-layout through a per-wave LDS staging buffer.
// ---------------------------------------------------------------------------
__global__ __launch_bounds__(128) void flash_attn_wmma(
    const __bf16* __restrict__ q, const __bf16* __restrict__ k,
    const __bf16* __restrict__ vt, __bf16* __restrict__ ctx)
{
    __shared__ __attribute__((aligned(32))) __bf16 Ks[2][32][64];  // 2 x 4 KB
    __shared__ __attribute__((aligned(32))) __bf16 Vs[2][64][32];  // 2 x 4 KB
    __shared__ __attribute__((aligned(32))) __bf16 Ps[4][16][32];  // 4 x 1 KB

    const int tid  = threadIdx.x;
    const int wave = tid >> 5;
    const int lane = tid & 31;
    const int half = lane >> 4;
    const int lr   = lane & 15;

    const int b  = blockIdx.z;
    const int h  = blockIdx.y;
    const int q0 = blockIdx.x * 64 + wave * 16;

    const size_t headRow = ((size_t)(b * NHEADS + h)) * SEQ;
    const size_t headD   = ((size_t)(b * NHEADS + h)) * HDIM;

    // per-thread stage-copy assignment: 2x K + 2x V async b128 per stage
    auto issue_kv = [&](int key0, int buf) {
#pragma unroll
        for (int i = 0; i < 2; ++i) {
            const int idx = tid + i * 128;                 // 0..255
            const int kr = idx >> 3, kc = (idx & 7) * 8;   // K tile: 32 x 64
            cp16_async(&Ks[buf][kr][kc], k + (headRow + key0 + kr) * HDIM + kc);
            const int vr = idx >> 2, vc = (idx & 3) * 8;   // V tile: 64 x 32
            cp16_async(&Vs[buf][vr][vc], vt + (headD + vr) * SEQ + key0 + vc);
        }
    };

    // Q fragments (registers for the whole loop): features 0..31 / 32..63
    const __bf16* qbase = q + (headRow + q0 + lr) * HDIM;
    const v16bf qa0 = load_a16x32(qbase + half * 16);
    const v16bf qa1 = load_a16x32(qbase + 32 + half * 16);

    v8f o[4] = {};
    float mrun[8], lrun[8];
#pragma unroll
    for (int j = 0; j < 8; ++j) { mrun[j] = -1.0e30f; lrun[j] = 0.0f; }

    issue_kv(0, 0);
    for (int it = 0; it < SEQ / 32; ++it) {
        const int buf = it & 1;
        if ((it + 1) * 32 < SEQ) {
            issue_kv((it + 1) * 32, buf ^ 1);
            WAIT_ASYNC(4);
        } else {
            WAIT_ASYNC(0);
        }
        __syncthreads();

        // K^T B-fragments from LDS: lane = key, 16 contiguous features
        const v16bf kb0 = *(const v16bf*)&Ks[buf][lr]     [half * 16];
        const v16bf kb1 = *(const v16bf*)&Ks[buf][lr]     [32 + half * 16];
        const v16bf kb2 = *(const v16bf*)&Ks[buf][16 + lr][half * 16];
        const v16bf kb3 = *(const v16bf*)&Ks[buf][16 + lr][32 + half * 16];

        v8f s0 = {}, s1 = {};
        s0 = WMMA_BF16(qa0, kb0, s0);
        s0 = WMMA_BF16(qa1, kb1, s0);
        s1 = WMMA_BF16(qa0, kb2, s1);
        s1 = WMMA_BF16(qa1, kb3, s1);

        // online softmax per row (row m = j + half*8 lives across 16 lanes)
#pragma unroll
        for (int j = 0; j < 8; ++j) {
            float t = fmaxf(s0[j], s1[j]);
#pragma unroll
            for (int m = 8; m >= 1; m >>= 1) t = fmaxf(t, __shfl_xor(t, m, 16));
            const float mn = fmaxf(mrun[j], t);
            const float al = __expf(mrun[j] - mn);
            const float p0 = __expf(s0[j] - mn);
            const float p1 = __expf(s1[j] - mn);
            float rs = p0 + p1;
#pragma unroll
            for (int m = 8; m >= 1; m >>= 1) rs += __shfl_xor(rs, m, 16);
            lrun[j] = lrun[j] * al + rs;
            mrun[j] = mn;
#pragma unroll
            for (int c = 0; c < 4; ++c) o[c][j] *= al;
            const int mrow = j + half * 8;
            Ps[wave][mrow][lr]      = (__bf16)p0;
            Ps[wave][mrow][16 + lr] = (__bf16)p1;
        }
        asm volatile("s_wait_dscnt 0" ::: "memory");  // wave-local LDS RAW

        // P as A-fragment (16 queries x 32 keys)
        const v16bf pa = load_a16x32(&Ps[wave][lr][half * 16]);

        // V B-fragments from LDS (transposed V): lane = d, contiguous keys
#pragma unroll
        for (int c = 0; c < 4; ++c) {
            const v16bf vb = *(const v16bf*)&Vs[buf][c * 16 + lr][half * 16];
            o[c] = WMMA_BF16(pa, vb, o[c]);
        }
        __syncthreads();
    }

    // normalize and write context in [B,S,H,D] (== [B,S,C]) layout, bf16
#pragma unroll
    for (int j = 0; j < 8; ++j) {
        const float inv = 1.0f / lrun[j];
        const int srow = q0 + j + half * 8;
        const size_t ob = (((size_t)(b * SEQ + srow)) * NHEADS + h) * HDIM;
#pragma unroll
        for (int c = 0; c < 4; ++c)
            ctx[ob + c * 16 + lr] = (__bf16)(o[c][j] * inv);
    }
}

// ---------------------------------------------------------------------------
// Launch
// ---------------------------------------------------------------------------
extern "C" void kernel_launch(void* const* d_in, const int* in_sizes, int n_in,
                              void* d_out, int out_size, void* d_ws, size_t ws_size,
                              hipStream_t stream) {
    const float* x       = (const float*)d_in[0];
    const float* w_qkv   = (const float*)d_in[1];
    const float* w_proj  = (const float*)d_in[2];
    const float* b_proj  = (const float*)d_in[3];
    const float* q_gamma = (const float*)d_in[4];
    const float* q_beta  = (const float*)d_in[5];
    const float* k_gamma = (const float*)d_in[6];
    const float* k_beta  = (const float*)d_in[7];

    char* w = (char*)d_ws;
    float*  qkv    = (float*)w;  w += (size_t)ROWS * QKV_O * 4;          // 48 MB
    __bf16* xb     = (__bf16*)w; w += (size_t)ROWS * DIMC * 2;           // 8 MB
    __bf16* wqkvb  = (__bf16*)w; w += (size_t)QKV_O * DIMC * 2;          // 6 MB
    __bf16* wprojb = (__bf16*)w; w += (size_t)DIMC * DIMC * 2;           // 2 MB
    __bf16* qb     = (__bf16*)w; w += (size_t)ROWS * DIMC * 2;           // 8 MB
    __bf16* kb     = (__bf16*)w; w += (size_t)ROWS * DIMC * 2;           // 8 MB
    __bf16* vtb    = (__bf16*)w; w += (size_t)ROWS * DIMC * 2;           // 8 MB
    __bf16* ctxb   = (__bf16*)w;                                         // 8 MB

    // 0) convert inputs/weights to bf16
    cvt_f32_to_bf16<<<(ROWS * DIMC + 255) / 256, 256, 0, stream>>>(x, xb, ROWS * DIMC);
    cvt_f32_to_bf16<<<(QKV_O * DIMC + 255) / 256, 256, 0, stream>>>(w_qkv, wqkvb, QKV_O * DIMC);
    cvt_f32_to_bf16<<<(DIMC * DIMC + 255) / 256, 256, 0, stream>>>(w_proj, wprojb, DIMC * DIMC);

    // 1) qkv = x @ w_qkv^T   (M=4096, N=3072, K=1024); block tile 64x128
    gemm_bf16_wmma<<<(ROWS / 64) * (QKV_O / 128), 256, 0, stream>>>(
        xb, wqkvb, qkv, nullptr, ROWS, QKV_O, DIMC);

    // 2) LN + RoPE + repack
    lnrope_pack<<<(BATCH * SEQ * NHEADS) / 8, 256, 0, stream>>>(
        qkv, q_gamma, q_beta, k_gamma, k_beta, qb, kb, vtb);

    // 3) attention
    flash_attn_wmma<<<dim3(SEQ / 64, NHEADS, BATCH), 128, 0, stream>>>(qb, kb, vtb, ctxb);

    // 4) out = ctx @ w_proj^T + b_proj  (M=4096, N=1024, K=1024) -> f32
    gemm_bf16_wmma<<<(ROWS / 64) * (DIMC / 128), 256, 0, stream>>>(
        ctxb, wprojb, (float*)d_out, b_proj, ROWS, DIMC, DIMC);
}
